// AKConv_52501680226813
// MI455X (gfx1250) — matrix operand order repaired and checked by reading, not compiled, not run on hardware
//
#include <hip/hip_runtime.h>

// AKConv fused for gfx1250 (MI455X):
//   offset conv -> bilinear gather (LDS, double-buffered)
//   -> f16 WMMA GEMM with A tiles staged by the Tensor Data Mover (TDM)
//   -> BN + SiLU.
//
// Workspace layout (requires ws_size >= ~30.5 MB):
//   [0, 26214400)            xT   : x transposed to (B, H*W, C) f32
//   [26214400, 29900800)     off  : offset conv output (B, 18, H*W) f32
//   [29900800, 30490624)     Wh   : w_conv converted to f16 (256 x 1152)

typedef _Float16 half_t;
typedef __attribute__((ext_vector_type(16))) _Float16 v16h;
typedef __attribute__((ext_vector_type(8)))  _Float16 v8h;
typedef __attribute__((ext_vector_type(8)))  float    v8f;
typedef __attribute__((ext_vector_type(4)))  unsigned int v4u;
typedef __attribute__((ext_vector_type(8)))  int      v8i;
typedef __attribute__((ext_vector_type(4)))  int      v4i;

#define BB   8
#define CIN  128
#define OC   256
#define NPT  9
#define HH   80
#define WWD  80
#define HW   6400
#define KTOT 1152
#define NPIX 51200
#define NCHUNK (KTOT / 32)
#define EPSV 1e-5f

// ---------------- prep: x (NCHW) -> xT (N, HW, C) ----------------
__global__ void k_transpose(const float* __restrict__ x, float* __restrict__ xT) {
  int o = blockIdx.x * blockDim.x + threadIdx.x;
  if (o >= BB * HW * CIN) return;
  int c  = o % CIN;
  int hw = (o / CIN) % HW;
  int b  = o / (CIN * HW);
  xT[o] = x[(b * CIN + c) * HW + hw];
}

// ---------------- prep: w_conv f32 -> f16 ----------------
__global__ void k_wcvt(const float* __restrict__ w, half_t* __restrict__ wh) {
  int i = blockIdx.x * blockDim.x + threadIdx.x;
  if (i < OC * KTOT) wh[i] = (half_t)w[i];
}

// ---------------- offset conv: 3x3, 128 -> 18, pad 1 ----------------
__global__ void k_offconv(const float* __restrict__ x, const float* __restrict__ w_off,
                          const float* __restrict__ b_off, float* __restrict__ off) {
  int bh = blockIdx.x;            // 0 .. B*H-1
  int b = bh / HH, h = bh % HH;
  for (int idx = threadIdx.x; idx < 2 * NPT * WWD; idx += blockDim.x) {
    int ch = idx / WWD, w = idx % WWD;
    float acc = b_off[ch];
    for (int di = -1; di <= 1; ++di) {
      int hh = h + di;
      if (hh < 0 || hh >= HH) continue;
      for (int dj = -1; dj <= 1; ++dj) {
        int ww = w + dj;
        if (ww < 0 || ww >= WWD) continue;
        const float* xp = x + (b * CIN) * HW + hh * WWD + ww;
        const float* wp = w_off + ((ch * CIN) * 3 + (di + 1)) * 3 + (dj + 1);
        #pragma unroll 4
        for (int c = 0; c < CIN; ++c)
          acc += xp[c * HW] * wp[c * 9];
      }
    }
    off[(b * 18 + ch) * HW + h * WWD + w] = acc;
  }
}

// Issue one TDM load of a (256 rows x 32 cols) f16 tile of Wh (256 x 1152,
// row-major) starting at column k0, into LDS at lds_addr.  2-D descriptor:
// groups 2/3 (and the extra descriptor words) zero.  data_size=2B,
// dims/strides in elements, addresses in bytes.
__device__ __forceinline__ void tdm_load_A(const half_t* Wh, int k0, unsigned lds_addr) {
  unsigned long long ga = (unsigned long long)(uintptr_t)Wh + (unsigned long long)k0 * 2u;
  v4u g0 = { 1u,                       // count=1, user mode, no gather
             lds_addr,                 // lds_addr [63:32]
             (unsigned)ga,             // global_addr low   [95:64]
             (unsigned)(ga >> 32) | (2u << 30) };  // addr hi + type=2 [127:96]
  v8i g1 = { 0x00010000,               // data_size=1 (2 bytes), mask=0
             (int)(KTOT << 16),        // tensor_dim0[15:0]=1152 at [63:48]
             (int)(OC << 16),          // dim0 hi=0; tensor_dim1[15:0]=256 at [95:80]
             (int)(32u << 16),         // dim1 hi=0; tile_dim0=32 at [127:112]
             (int)OC,                  // tile_dim1=256; tile_dim2=0
             (int)KTOT,                // tensor_dim0_stride lo = 1152
             0, 0 };                   // stride hi, tensor_dim1_stride (unused)
  v4i gz4 = { 0, 0, 0, 0 };
  v8i gz8 = { 0, 0, 0, 0, 0, 0, 0, 0 };
  __builtin_amdgcn_tensor_load_to_lds(g0, g1, gz4, gz4, gz8, 0);
}

// ---------------- main: gather + TDM-staged WMMA GEMM + BN + SiLU ----------------
// Block = 256 threads = 8 waves: 4 M-groups (64 rows each) x 2 N-groups (16 px).
// Block tile: M = 256 (all out channels), N = 32 pixels, K in 36 chunks of 32.
// Double-buffered: A tile (TDM -> LDS), B tile (bilinear gather -> LDS).
__launch_bounds__(256)
__global__ void k_main(const float* __restrict__ xT, const float* __restrict__ off,
                       const half_t* __restrict__ Wh,
                       const float* __restrict__ bn_g, const float* __restrict__ bn_b,
                       const float* __restrict__ bn_mu, const float* __restrict__ bn_var,
                       float* __restrict__ y) {
  __shared__ __align__(32) half_t s_at[2][OC][32];  // A tiles: 2 x 16 KB
  __shared__ __align__(32) half_t s_bt[2][32][32];  // B tiles: 2 x 2 KB

  const int tid  = threadIdx.x;
  const int lane = tid & 31;
  const int wave = tid >> 5;
  const int mgrp = wave >> 1;    // 0..3 -> M base = mgrp*64
  const int ngrp = wave & 1;     // 0..1 -> N base = ngrp*16
  const int n0   = blockIdx.x * 32;
  const int sel  = lane >> 4;    // half-wave select
  const int mrow = lane & 15;

  // ---- per-thread gather role: pixel j = tid>>3, 4 channels at (tid&7)*4 ----
  const int gj   = tid >> 3;
  const int gkk  = (tid & 7) * 4;
  const int gpix = n0 + gj;
  const int gb   = gpix / HW, ghw = gpix % HW;
  const int gh   = ghw / WWD, gw = ghw % WWD;

  int   cb[4];   // bilinear corner base offsets into xT (elements)
  float cw[4];   // bilinear corner weights (0 if out of image)

  auto compute_meta = [&](int pt) {
    float ox = off[(gb * 18 + pt) * HW + ghw];
    float oy = off[(gb * 18 + NPT + pt) * HW + ghw];
    float px = (float)gh + (float)(pt / 3 - 1) + ox;   // rows = x
    float py = (float)gw + (float)(pt % 3 - 1) + oy;   // cols = y
    float fx = floorf(px), fy = floorf(py);
    float dx = px - fx, dy = py - fy;
    int x0 = (int)fx, y0 = (int)fy;
    #pragma unroll
    for (int ci = 0; ci < 4; ++ci) {
      int xi = x0 + (ci >> 1), yi = y0 + (ci & 1);
      float wx = (ci >> 1) ? dx : 1.f - dx;
      float wy = (ci & 1) ? dy : 1.f - dy;
      bool valid = (xi >= 0) && (xi < HH) && (yi >= 0) && (yi < WWD);
      int xc = min(max(xi, 0), HH - 1);
      int yc = min(max(yi, 0), WWD - 1);
      cb[ci] = (gb * HW + xc * WWD + yc) * CIN;
      cw[ci] = valid ? wx * wy : 0.f;
    }
  };

  auto gather = [&](int k0, int nb) {
    int c = (k0 & (CIN - 1)) + gkk;
    float4 r = make_float4(0.f, 0.f, 0.f, 0.f);
    #pragma unroll
    for (int ci = 0; ci < 4; ++ci) {
      float wgt = cw[ci];
      const float4 v = *(const float4*)(xT + cb[ci] + c);
      r.x += wgt * v.x; r.y += wgt * v.y; r.z += wgt * v.z; r.w += wgt * v.w;
    }
    half_t* dst = &s_bt[nb][gj][gkk];
    dst[0] = (half_t)r.x; dst[1] = (half_t)r.y;
    dst[2] = (half_t)r.z; dst[3] = (half_t)r.w;
  };

  v8f acc[4] = {};               // 4 M-subtiles of 16x16 f32

  // ---- prologue: stage chunk 0 (A via TDM, B via gather) into buffer 0 ----
  if (wave == 0)
    tdm_load_A(Wh, 0, (unsigned)(uintptr_t)&s_at[0][0][0]);
  compute_meta(0);
  gather(0, 0);
  if (wave == 0)
    __builtin_amdgcn_s_wait_tensorcnt(0);
  __syncthreads();

  for (int i = 0; i < NCHUNK; ++i) {
    const int buf = i & 1;
    const int k0  = i * 32;

    // Prefetch next A tile via TDM into the other buffer (wave 0 only; one
    // DMA per workgroup — EXEC is ignored by tensor ops).
    if (i + 1 < NCHUNK && wave == 0)
      tdm_load_A(Wh, k0 + 32, (unsigned)(uintptr_t)&s_at[buf ^ 1][0][0]);

    // B fragment (32x16 f16): lanes 0-15 hold K 0..15, lanes 16-31 K 16..31.
    v16h bfrag = *(const v16h*)&s_bt[buf][ngrp * 16 + mrow][sel * 16];

    // Issue ALL four A-fragment loads (8 x ds_load_b128) before any WMMA so
    // the waits can be batched instead of one full s_wait_dscnt per MMA.
    // A fragment (16x32 f16) from LDS, same K order as B fragment:
    //   lanes 0-15:  VGPR0-3 = K+0..7,   VGPR4-7 = K+8..15
    //   lanes 16-31: VGPR0-3 = K+16..23, VGPR4-7 = K+24..31
    v16h af[4];
    #pragma unroll
    for (int s = 0; s < 4; ++s) {
      int m = mgrp * 64 + s * 16 + mrow;
      v8h alo = *(const v8h*)&s_at[buf][m][sel * 16];
      v8h ahi = *(const v8h*)&s_at[buf][m][sel * 16 + 8];
      #pragma unroll
      for (int e = 0; e < 8; ++e) { af[s][e] = alo[e]; af[s][e + 8] = ahi[e]; }
    }

    // Gather next B tile into the other buffer (overlaps the WMMAs below).
    if (i + 1 < NCHUNK) {
      int k1 = k0 + 32;
      if ((k1 & (CIN - 1)) == 0) compute_meta(k1 >> 7);
      gather(k1, buf ^ 1);
    }

    #pragma unroll
    for (int s = 0; s < 4; ++s)
      acc[s] = __builtin_amdgcn_wmma_f32_16x16x32_f16(
          false, af[s], false, bfrag, (short)0, acc[s], false, false);

    // Publish next buffers: wave 0 ensures its TDM finished, then barrier.
    if (wave == 0)
      __builtin_amdgcn_s_wait_tensorcnt(0);
    __syncthreads();
  }

  // Epilogue: BN (eval) + SiLU, store NCHW.
  // D layout: VGPR r -> M = r (lanes 0-15) / r+8 (lanes 16-31); col N = lane&15.
  int pixel = n0 + ngrp * 16 + mrow;
  int b = pixel / HW, hw = pixel % HW;
  #pragma unroll
  for (int s = 0; s < 4; ++s) {
    int mbase = mgrp * 64 + s * 16 + sel * 8;
    #pragma unroll
    for (int r = 0; r < 8; ++r) {
      int o = mbase + r;
      float sc  = bn_g[o] * rsqrtf(bn_var[o] + EPSV);
      float val = (acc[s][r] - bn_mu[o]) * sc + bn_b[o];
      float out = val / (1.f + __expf(-val));
      y[(b * OC + o) * HW + hw] = out;
    }
  }
}

extern "C" void kernel_launch(void* const* d_in, const int* in_sizes, int n_in,
                              void* d_out, int out_size, void* d_ws, size_t ws_size,
                              hipStream_t stream) {
  const float* x      = (const float*)d_in[0];
  const float* w_off  = (const float*)d_in[1];
  const float* b_off  = (const float*)d_in[2];
  const float* w_conv = (const float*)d_in[3];
  const float* bn_g   = (const float*)d_in[4];
  const float* bn_b   = (const float*)d_in[5];
  const float* bn_mu  = (const float*)d_in[6];
  const float* bn_var = (const float*)d_in[7];
  float* y = (float*)d_out;

  char*   ws  = (char*)d_ws;
  float*  xT  = (float*)ws;                                  // 26,214,400 B
  float*  off = (float*)(ws + 26214400);                     //  3,686,400 B
  half_t* Wh  = (half_t*)(ws + 26214400 + 3686400);          //    589,824 B

  k_transpose<<<(BB * HW * CIN + 255) / 256, 256, 0, stream>>>(x, xT);
  k_wcvt<<<(OC * KTOT + 255) / 256, 256, 0, stream>>>(w_conv, Wh);
  k_offconv<<<BB * HH, 256, 0, stream>>>(x, w_off, b_off, off);
  k_main<<<NPIX / 32, 256, 0, stream>>>(xT, off, Wh, bn_g, bn_b, bn_mu, bn_var, y);
}
